// SingleSourceGNN_1039382085723
// MI455X (gfx1250) — compile-verified, matrix operand order checked by reading, not run on hardware
//
#include <hip/hip_runtime.h>
#include <hip/hip_bf16.h>

typedef __attribute__((ext_vector_type(16))) _Float16 v16h;
typedef __attribute__((ext_vector_type(8)))  float    v8f;

#define HEADS 4
#define CCH   16
#define OUTCH 64
#define NEG_SLOPE 0.2f

// ---------------------------------------------------------------------------
// WMMA GEMM: h[n,64] = x[n,K] @ W[K,64]   (K = 128 or 64, n multiple of 16)
// One wave handles a 16-row tile and all four 16-col tiles; K chunked by 32.
// A fragment (16x32 f16): lane l -> row (l&15); K pairs per VGPR with the
// lane>=16 half holding K offsets +8 (per CDNA5 ISA 7.12.2). B mirrors it
// with N=(l&15). C/D: N=(l&15), M = 8*(l>>4) + vgpr.
// ---------------------------------------------------------------------------
__global__ __launch_bounds__(256) void gat_gemm_wmma(
    const float* __restrict__ x, const float* __restrict__ W,
    float* __restrict__ h, int nrows, int K)
{
    const int lane  = threadIdx.x & 31;
    const int wave  = threadIdx.x >> 5;
    const int mTile = blockIdx.x * 8 + wave;
    if (mTile * 16 >= nrows) return;          // wave-uniform: EXEC stays all-1s

    const int row  = mTile * 16 + (lane & 15);
    const int koff = (lane >> 4) * 8;         // lane-half K offset
    const int col  = lane & 15;

    v8f acc0 = {}, acc1 = {}, acc2 = {}, acc3 = {};

    for (int kc = 0; kc < K; kc += 32) {
        // ---- A fragment: 16 f16 per lane (K = kc+koff+0..7 and kc+16+koff+0..7)
        v16h a;
        {
            const float4* xr0 = (const float4*)(x + (size_t)row * K + kc + koff);
            const float4* xr1 = (const float4*)(x + (size_t)row * K + kc + 16 + koff);
            float4 p0 = xr0[0], p1 = xr0[1];
            float4 q0 = xr1[0], q1 = xr1[1];
            a[0]  = (_Float16)p0.x; a[1]  = (_Float16)p0.y;
            a[2]  = (_Float16)p0.z; a[3]  = (_Float16)p0.w;
            a[4]  = (_Float16)p1.x; a[5]  = (_Float16)p1.y;
            a[6]  = (_Float16)p1.z; a[7]  = (_Float16)p1.w;
            a[8]  = (_Float16)q0.x; a[9]  = (_Float16)q0.y;
            a[10] = (_Float16)q0.z; a[11] = (_Float16)q0.w;
            a[12] = (_Float16)q1.x; a[13] = (_Float16)q1.y;
            a[14] = (_Float16)q1.z; a[15] = (_Float16)q1.w;
        }
        // ---- 4 column tiles of W
        #pragma unroll
        for (int nt = 0; nt < 4; ++nt) {
            v16h b;
            const float* wp0 = W + (size_t)(kc + koff)      * OUTCH + nt * 16 + col;
            const float* wp1 = W + (size_t)(kc + 16 + koff) * OUTCH + nt * 16 + col;
            #pragma unroll
            for (int i = 0; i < 8; ++i) b[i]     = (_Float16)wp0[(size_t)i * OUTCH];
            #pragma unroll
            for (int i = 0; i < 8; ++i) b[8 + i] = (_Float16)wp1[(size_t)i * OUTCH];

            v8f c = (nt == 0) ? acc0 : (nt == 1) ? acc1 : (nt == 2) ? acc2 : acc3;
            c = __builtin_amdgcn_wmma_f32_16x16x32_f16(
                    /*neg_a=*/false, a, /*neg_b=*/false, b,
                    /*c_mod=*/(short)0, c, /*reuse_a=*/false, /*reuse_b=*/false);
            if (nt == 0) acc0 = c; else if (nt == 1) acc1 = c;
            else if (nt == 2) acc2 = c; else acc3 = c;
        }
    }

    // ---- store D: row = mTile*16 + 8*(lane>>4) + j ; col = nt*16 + (lane&15)
    const int mrow0 = mTile * 16 + (lane >> 4) * 8;
    #pragma unroll
    for (int j = 0; j < 8; ++j) {
        float* hr = h + (size_t)(mrow0 + j) * OUTCH + (lane & 15);
        hr[0]  = acc0[j];
        hr[16] = acc1[j];
        hr[32] = acc2[j];
        hr[48] = acc3[j];
    }
}

// ---------------------------------------------------------------------------
// Per (node, head): a_s/a_d dots, init segment-max=-inf, denom=0, outacc=0.
// ---------------------------------------------------------------------------
__global__ void gat_prep(const float* __restrict__ h,
                         const float* __restrict__ att_src,
                         const float* __restrict__ att_dst,
                         float* __restrict__ a_s, float* __restrict__ a_d,
                         float* __restrict__ mbuf, float* __restrict__ denom,
                         float* __restrict__ outacc, int n)
{
    int idx = blockIdx.x * blockDim.x + threadIdx.x;   // node*4 + head
    if (idx >= n * HEADS) return;
    int node = idx >> 2, hd = idx & 3;
    const float4* hp = (const float4*)(h + (size_t)node * OUTCH + hd * CCH);
    const float4* av = (const float4*)(att_src + hd * CCH);
    const float4* bv = (const float4*)(att_dst + hd * CCH);
    float4*       op = (float4*)(outacc + (size_t)node * OUTCH + hd * CCH);
    float ss = 0.f, dd = 0.f;
    #pragma unroll
    for (int i = 0; i < 4; ++i) {
        float4 hv = hp[i], a4 = av[i], b4 = bv[i];
        ss += hv.x * a4.x + hv.y * a4.y + hv.z * a4.z + hv.w * a4.w;
        dd += hv.x * b4.x + hv.y * b4.y + hv.z * b4.z + hv.w * b4.w;
        op[i] = make_float4(0.f, 0.f, 0.f, 0.f);
    }
    a_s[idx]   = ss;
    a_d[idx]   = dd;
    mbuf[idx]  = -__builtin_inff();
    denom[idx] = 0.f;
}

// float atomic max via signed-max (v>=0) / unsigned-min (v<0) bit ordering.
__device__ inline void atomicMaxF(float* addr, float v) {
    if (v >= 0.f) atomicMax((int*)addr, __float_as_int(v));
    else          atomicMin((unsigned int*)addr, __float_as_uint(v));
}

__device__ inline void edge_nodes(const long long* __restrict__ ei, long long E,
                                  long long e, int& s, int& d) {
    if (e < E) { s = (int)ei[e]; d = (int)ei[E + e]; }
    else       { s = d = (int)(e - E); }                // self loop
}

// ---------------------------------------------------------------------------
// Pass 1: segment max of leaky_relu(a_s[src]+a_d[dst]) over incoming edges.
// ---------------------------------------------------------------------------
__global__ void gat_edge_max(const long long* __restrict__ ei, long long E, int n,
                             const float* __restrict__ a_s,
                             const float* __restrict__ a_d,
                             float* __restrict__ mbuf)
{
    long long e = (long long)blockIdx.x * blockDim.x + threadIdx.x;
    if (e >= E + n) return;
    int s, d; edge_nodes(ei, E, e, s, d);
    float4 as = *(const float4*)(a_s + (size_t)s * HEADS);
    float4 ad = *(const float4*)(a_d + (size_t)d * HEADS);
    float v0 = as.x + ad.x, v1 = as.y + ad.y, v2 = as.z + ad.z, v3 = as.w + ad.w;
    v0 = v0 > 0.f ? v0 : NEG_SLOPE * v0;
    v1 = v1 > 0.f ? v1 : NEG_SLOPE * v1;
    v2 = v2 > 0.f ? v2 : NEG_SLOPE * v2;
    v3 = v3 > 0.f ? v3 : NEG_SLOPE * v3;
    float* mp = mbuf + (size_t)d * HEADS;
    atomicMaxF(mp + 0, v0); atomicMaxF(mp + 1, v1);
    atomicMaxF(mp + 2, v2); atomicMaxF(mp + 3, v3);
}

// ---------------------------------------------------------------------------
// Pass 2: denom[dst] += exp(e - m[dst])
// ---------------------------------------------------------------------------
__global__ void gat_edge_sum(const long long* __restrict__ ei, long long E, int n,
                             const float* __restrict__ a_s,
                             const float* __restrict__ a_d,
                             const float* __restrict__ mbuf,
                             float* __restrict__ denom)
{
    long long e = (long long)blockIdx.x * blockDim.x + threadIdx.x;
    if (e >= E + n) return;
    int s, d; edge_nodes(ei, E, e, s, d);
    float4 as = *(const float4*)(a_s  + (size_t)s * HEADS);
    float4 ad = *(const float4*)(a_d  + (size_t)d * HEADS);
    float4 mv = *(const float4*)(mbuf + (size_t)d * HEADS);
    float v0 = as.x + ad.x, v1 = as.y + ad.y, v2 = as.z + ad.z, v3 = as.w + ad.w;
    v0 = v0 > 0.f ? v0 : NEG_SLOPE * v0;
    v1 = v1 > 0.f ? v1 : NEG_SLOPE * v1;
    v2 = v2 > 0.f ? v2 : NEG_SLOPE * v2;
    v3 = v3 > 0.f ? v3 : NEG_SLOPE * v3;
    float* dp = denom + (size_t)d * HEADS;
    atomicAdd(dp + 0, __expf(v0 - mv.x));
    atomicAdd(dp + 1, __expf(v1 - mv.y));
    atomicAdd(dp + 2, __expf(v2 - mv.z));
    atomicAdd(dp + 3, __expf(v3 - mv.w));
}

// ---------------------------------------------------------------------------
// Pass 3: outacc[dst, head, :] += alpha * h[src, head, :]
// One thread per (edge, head): 4x float4 gather + 16 atomic adds (L2 RMW).
// ---------------------------------------------------------------------------
__global__ void gat_edge_scatter(const long long* __restrict__ ei, long long E, int n,
                                 const float* __restrict__ a_s,
                                 const float* __restrict__ a_d,
                                 const float* __restrict__ mbuf,
                                 const float* __restrict__ denom,
                                 const float* __restrict__ h,
                                 float* __restrict__ outacc)
{
    long long idx = (long long)blockIdx.x * blockDim.x + threadIdx.x;
    if (idx >= (E + n) * HEADS) return;
    long long e = idx >> 2;
    int hd = (int)(idx & 3);
    int s, d; edge_nodes(ei, E, e, s, d);
    float ev = a_s[(size_t)s * HEADS + hd] + a_d[(size_t)d * HEADS + hd];
    ev = ev > 0.f ? ev : NEG_SLOPE * ev;
    float alpha = __expf(ev - mbuf[(size_t)d * HEADS + hd]) /
                  (denom[(size_t)d * HEADS + hd] + 1e-16f);
    const float4* hp = (const float4*)(h + (size_t)s * OUTCH + hd * CCH);
    float*        op = outacc + (size_t)d * OUTCH + hd * CCH;
    #pragma unroll
    for (int i = 0; i < 4; ++i) {
        float4 hv = hp[i];
        atomicAdd(op + 4 * i + 0, hv.x * alpha);
        atomicAdd(op + 4 * i + 1, hv.y * alpha);
        atomicAdd(op + 4 * i + 2, hv.z * alpha);
        atomicAdd(op + 4 * i + 3, hv.w * alpha);
    }
}

// ---------------------------------------------------------------------------
// Finalize: y = outacc + bias, optional ReLU.
// ---------------------------------------------------------------------------
__global__ void gat_finalize(const float* __restrict__ outacc,
                             const float* __restrict__ bias,
                             float* __restrict__ y, int n, int do_relu)
{
    long long idx = (long long)blockIdx.x * blockDim.x + threadIdx.x;
    if (idx >= (long long)n * OUTCH) return;
    float v = outacc[idx] + bias[idx & (OUTCH - 1)];
    if (do_relu) v = fmaxf(v, 0.f);
    y[idx] = v;
}

// ---------------------------------------------------------------------------
extern "C" void kernel_launch(void* const* d_in, const int* in_sizes, int n_in,
                              void* d_out, int out_size, void* d_ws, size_t ws_size,
                              hipStream_t stream)
{
    const float*     x  = (const float*)d_in[0];
    const long long* ei = (const long long*)d_in[1];
    const int  n = in_sizes[0] / 128;          // 50000
    const long long E = in_sizes[1] / 2;       // 1.6M
    const long long tot = E + n;

    float* ws     = (float*)d_ws;
    float* xbuf   = ws;                              // n*64  next-layer input
    float* hbuf   = xbuf   + (size_t)n * OUTCH;      // n*64  h = xW
    float* outacc = hbuf   + (size_t)n * OUTCH;      // n*64  scatter accumulator
    float* a_s    = outacc + (size_t)n * OUTCH;      // n*4
    float* a_d    = a_s    + (size_t)n * HEADS;      // n*4
    float* mbuf   = a_d    + (size_t)n * HEADS;      // n*4
    float* denom  = mbuf   + (size_t)n * HEADS;      // n*4

    const float* cur = x;
    int K = 128;
    for (int layer = 0; layer < 3; ++layer) {
        const float* W    = (const float*)d_in[2 + layer * 4 + 0];
        const float* asw  = (const float*)d_in[2 + layer * 4 + 1];
        const float* adw  = (const float*)d_in[2 + layer * 4 + 2];
        const float* bias = (const float*)d_in[2 + layer * 4 + 3];

        unsigned mtiles = (unsigned)((n + 15) / 16);
        gat_gemm_wmma<<<dim3((mtiles + 7) / 8), dim3(256), 0, stream>>>(cur, W, hbuf, n, K);

        unsigned pgrid = (unsigned)((n * HEADS + 255) / 256);
        gat_prep<<<dim3(pgrid), dim3(256), 0, stream>>>(hbuf, asw, adw, a_s, a_d,
                                                        mbuf, denom, outacc, n);

        unsigned egrid = (unsigned)((tot + 255) / 256);
        gat_edge_max<<<dim3(egrid), dim3(256), 0, stream>>>(ei, E, n, a_s, a_d, mbuf);
        gat_edge_sum<<<dim3(egrid), dim3(256), 0, stream>>>(ei, E, n, a_s, a_d, mbuf, denom);

        unsigned sgrid = (unsigned)((tot * HEADS + 255) / 256);
        gat_edge_scatter<<<dim3(sgrid), dim3(256), 0, stream>>>(ei, E, n, a_s, a_d,
                                                                mbuf, denom, hbuf, outacc);

        float* y = (layer < 2) ? xbuf : (float*)d_out;
        unsigned fgrid = (unsigned)(((long long)n * OUTCH + 255) / 256);
        gat_finalize<<<dim3(fgrid), dim3(256), 0, stream>>>(outacc, bias, y, n, layer < 2);

        cur = xbuf;
        K = OUTCH;
    }
}
